// HardAttention_70841190580339
// MI455X (gfx1250) — matrix operand order repaired and checked by reading, compiled, not run on hardware
//
#include <hip/hip_runtime.h>
#include <hip/hip_bf16.h>
#include <math.h>

#define B_ 128
#define L_ 4096
#define D_ 128
#define U_ 128
#define H_ 512

typedef __attribute__((ext_vector_type(16))) __bf16 v16bf;
typedef __attribute__((ext_vector_type(8)))  __bf16 v8bf;
typedef __attribute__((ext_vector_type(8)))  float  v8f;
typedef __attribute__((ext_vector_type(8)))  unsigned int v8u;
typedef __attribute__((ext_vector_type(4)))  unsigned int v4u;
typedef __attribute__((ext_vector_type(8)))  int v8i;
typedef __attribute__((ext_vector_type(4)))  int v4i;

// round-to-nearest-even f32 -> bf16 (used once per block for Wf)
__device__ __forceinline__ __bf16 f2bf(float f) {
    unsigned u = __float_as_uint(f);
    unsigned r = u + 0x7FFFu + ((u >> 16) & 1u);
    unsigned short h = (unsigned short)(r >> 16);
    return __builtin_bit_cast(__bf16, h);
}

// pack two f32 into a bf16 pair with one v_perm_b32 (truncating round)
__device__ __forceinline__ unsigned pkbf(float lo, float hi) {
    return __builtin_amdgcn_perm(__float_as_uint(hi), __float_as_uint(lo), 0x07060302u);
}

// ---------------------------------------------------------------------------
// Kernel 1: c[b][u] = bf[u] + bh[u] + hidden[b,:] . Wh[:,u]      (tiny GEMV)
// ---------------------------------------------------------------------------
__global__ __launch_bounds__(128) void hidproj_kernel(
    const float* __restrict__ hidden, const float* __restrict__ Wh,
    const float* __restrict__ bf, const float* __restrict__ bh,
    float* __restrict__ cvec) {
    const int b = blockIdx.x;
    const int u = threadIdx.x;
    const float* hr = hidden + (size_t)b * H_;
    float s = bf[u] + bh[u];
    for (int h = 0; h < H_; ++h)
        s = fmaf(hr[h], Wh[h * U_ + u], s);
    cvec[b * U_ + u] = s;
}

// ---------------------------------------------------------------------------
// Kernel 2: fused  logits[b,l] = Ws . tanh(features[b,l,:] @ Wf + c[b,:]) + bs
// TDM DMA of the 128x128 f32 feature tile into LDS (padded rows), then
// bf16 WMMA 16x16x32 with f32 accumulate. 8 waves/block, 16 rows per wave.
// ---------------------------------------------------------------------------
__global__ __launch_bounds__(256) void fused_logits_kernel(
    const float* __restrict__ features, const float* __restrict__ Wf,
    const float* __restrict__ cvec, const float* __restrict__ Ws,
    const float* __restrict__ bs, float* __restrict__ logits) {
    constexpr int WFS  = 136;  // Wf K-stride (elems): 272B = 17*16B, conflict-free
    constexpr int TROW = 132;  // tile row stride (dwords): 128 + 4 TDM pad dwords
    __shared__ __align__(16) float  sTile[128 * TROW];   // 67584 B, TDM destination
    __shared__ __align__(32) __bf16 sWfT[U_ * WFS];      // 34816 B, [n][k] bf16
    __shared__ float sC[U_];
    __shared__ float sWs[U_];

    const int tid  = threadIdx.x;
    const int b    = blockIdx.y;
    const int wave = tid >> 5;

    // ---- issue TDM: features[b, blockIdx.x*128 .. +127, :] -> sTile --------
    if (wave == 0) {
        unsigned long long gaddr = (unsigned long long)(uintptr_t)(
            features + ((size_t)b * L_ + (size_t)blockIdx.x * 128) * D_);
        unsigned ldsa = (unsigned)(uintptr_t)(void*)sTile;  // low 32 bits = LDS offset
        v4u g0;
        g0[0] = 1u;                                   // count=1, user descriptor
        g0[1] = ldsa;                                 // lds_addr
        g0[2] = (unsigned)(gaddr & 0xFFFFFFFFu);      // global_addr[31:0]
        g0[3] = (unsigned)((gaddr >> 32) & 0x1FFFFFFu) | (2u << 30); // addr[56:32], type=2
        v8u g1;
        g1[0] = (2u << 16)        // data_size = 4 bytes
              | (1u << 20)        // pad_enable
              | (6u << 22)        // pad_interval: every 128 dwords (512B row)
              | (3u << 25);       // pad_amount: 4 dwords (16B) -> 132-dword rows
        g1[1] = (128u << 16);     // tensor_dim0[15:0] = 128 (bits 63:48)
        g1[2] = 0u;               // tensor_dim0 hi = 0, tensor_dim1[15:0] = 0
        g1[3] = ((unsigned)((B_ * L_) >> 16))  // tensor_dim1[31:16] (rows total)
              | (128u << 16);     // tile_dim0 = 128
        g1[4] = 128u;             // tile_dim1 = 128, tile_dim2 = 0 (unused)
        g1[5] = 128u;             // tensor_dim0_stride[31:0] = 128 elems
        g1[6] = 0u;               // stride hi, tensor_dim1_stride lo
        g1[7] = 0u;
        v4i z4 = {0, 0, 0, 0};    // groups 2/3: dims beyond 2 unused
        v8i z8 = {0, 0, 0, 0, 0, 0, 0, 0};
        __builtin_amdgcn_tensor_load_to_lds(g0, __builtin_bit_cast(v8i, g1),
                                            z4, z4, z8, 0);
    }

    // ---- overlap: Wf [D][U] f32 (global) -> sWfT[u][d] bf16 (LDS) ----------
    for (int i = tid; i < D_ * U_; i += 256) {
        const int d = i >> 7;
        const int u = i & 127;
        sWfT[u * WFS + d] = f2bf(Wf[i]);
    }
    if (tid < U_) {
        sC[tid]  = cvec[b * U_ + tid];
        sWs[tid] = Ws[tid];
    }
    if (wave == 0) __builtin_amdgcn_s_wait_tensorcnt(0);
    __syncthreads();

    const int lane = tid & 31;
    const int half = lane >> 4;   // which 16-lane half
    const int l15  = lane & 15;
    const int rowBase = blockIdx.x * 128 + wave * 16;

    // A-matrix: lane (both halves) holds row M = l15; halves take different K chunks
    const float* frow = sTile + (wave * 16 + l15) * TROW;

    v8f acc[8] = {};  // 8 N-tiles of the 16x128 output

    #pragma unroll
    for (int k = 0; k < 4; ++k) {            // K steps of 32
        // A layout (16-bit, 16x32): lanes 0-15: K = 0..7 then 16..23
        //                           lanes 16-31: K = 8..15 then 24..31
        const int ka = 32 * k + 8 * half;
        float4 x0 = *(const float4*)(frow + ka);
        float4 x1 = *(const float4*)(frow + ka + 4);
        float4 x2 = *(const float4*)(frow + ka + 16);
        float4 x3 = *(const float4*)(frow + ka + 20);
        v8u ap;
        ap[0] = pkbf(x0.x, x0.y); ap[1] = pkbf(x0.z, x0.w);
        ap[2] = pkbf(x1.x, x1.y); ap[3] = pkbf(x1.z, x1.w);
        ap[4] = pkbf(x2.x, x2.y); ap[5] = pkbf(x2.z, x2.w);
        ap[6] = pkbf(x3.x, x3.y); ap[7] = pkbf(x3.z, x3.w);
        v16bf a = __builtin_bit_cast(v16bf, ap);

        // B layout (16-bit, 32x16): lane holds N = l15, K = 8*half..+7 then 16+8*half..+7
        const int kb = 32 * k + 8 * half;
        #pragma unroll
        for (int t = 0; t < 8; ++t) {
            const __bf16* bp = &sWfT[(t * 16 + l15) * WFS + kb];
            v8bf blo = *(const v8bf*)bp;
            v8bf bhi = *(const v8bf*)(bp + 16);
            v16bf bb = __builtin_shufflevector(blo, bhi,
                0, 1, 2, 3, 4, 5, 6, 7, 8, 9, 10, 11, 12, 13, 14, 15);
            acc[t] = __builtin_amdgcn_wmma_f32_16x16x32_bf16(
                false, a, false, bb, (short)0, acc[t], false, false);
        }
    }

    // C/D layout: VGPR r, lane<16 -> (M=r, N=l15+16t); lane>=16 -> (M=r+8, ...)
    float partial[8] = {0.f, 0.f, 0.f, 0.f, 0.f, 0.f, 0.f, 0.f};
    #pragma unroll
    for (int t = 0; t < 8; ++t) {
        const int n = t * 16 + l15;
        const float w = sWs[n];
        const float cadd = sC[n];
        #pragma unroll
        for (int r = 0; r < 8; ++r)
            partial[r] += w * tanhf(acc[t][r] + cadd);
    }
    // reduce across the 16 lanes of each half (masks 1,2,4,8 stay within a half)
    #pragma unroll
    for (int m = 1; m < 16; m <<= 1) {
        #pragma unroll
        for (int r = 0; r < 8; ++r)
            partial[r] += __shfl_xor(partial[r], m, 32);
    }

    if (l15 == 0) {
        const float bsv = bs[0];
        #pragma unroll
        for (int r = 0; r < 8; ++r)
            logits[(size_t)b * L_ + rowBase + r + 8 * half] = partial[r] + bsv;
    }
}

// ---------------------------------------------------------------------------
// Kernel 3: per-batch softmax over L (in place), greedy / Gumbel argmax,
// context gather. One block per batch row.
// ---------------------------------------------------------------------------
__global__ __launch_bounds__(256) void softmax_select_kernel(
    float* __restrict__ alpha,            // in: logits, out: softmax(alpha)
    const float* __restrict__ noise, const int* __restrict__ greedy,
    const float* __restrict__ features, float* __restrict__ ctx) {
    __shared__ float sv[256];
    __shared__ int si[256];
    __shared__ float s_max, s_sum;
    __shared__ int s_gloc, s_sloc;

    const int b = blockIdx.x;
    const int tid = threadIdx.x;
    float* lg = alpha + (size_t)b * L_;
    const float* nz = noise + (size_t)b * L_;

    float gval = -INFINITY; int gidx = 0;   // greedy argmax
    float stov = -INFINITY; int stoi = 0;   // gumbel-perturbed argmax
    for (int l = tid; l < L_; l += 256) {
        const float v = lg[l];
        if (v > gval) { gval = v; gidx = l; }
        const float g = -logf(-logf(nz[l] + 1e-9f) + 1e-9f);
        const float v2 = v + g;
        if (v2 > stov) { stov = v2; stoi = l; }
    }

    // argmax(logits) with first-index tie-break
    sv[tid] = gval; si[tid] = gidx; __syncthreads();
    for (int s = 128; s > 0; s >>= 1) {
        if (tid < s) {
            const float v2 = sv[tid + s]; const int i2 = si[tid + s];
            if (v2 > sv[tid] || (v2 == sv[tid] && i2 < si[tid])) { sv[tid] = v2; si[tid] = i2; }
        }
        __syncthreads();
    }
    if (tid == 0) { s_max = sv[0]; s_gloc = si[0]; }
    __syncthreads();

    // argmax(logits + gumbel)
    sv[tid] = stov; si[tid] = stoi; __syncthreads();
    for (int s = 128; s > 0; s >>= 1) {
        if (tid < s) {
            const float v2 = sv[tid + s]; const int i2 = si[tid + s];
            if (v2 > sv[tid] || (v2 == sv[tid] && i2 < si[tid])) { sv[tid] = v2; si[tid] = i2; }
        }
        __syncthreads();
    }
    if (tid == 0) s_sloc = si[0];
    __syncthreads();

    // sum of exp
    const float maxv = s_max;
    float accs = 0.f;
    for (int l = tid; l < L_; l += 256) accs += expf(lg[l] - maxv);
    sv[tid] = accs; __syncthreads();
    for (int s = 128; s > 0; s >>= 1) {
        if (tid < s) sv[tid] += sv[tid + s];
        __syncthreads();
    }
    if (tid == 0) s_sum = sv[0];
    __syncthreads();

    const float inv = 1.0f / s_sum;
    for (int l = tid; l < L_; l += 256) lg[l] = expf(lg[l] - maxv) * inv;

    // per-sample hard selection + context gather
    const int loc = greedy[b] ? s_gloc : s_sloc;
    if (tid < D_)
        ctx[b * D_ + tid] = features[((size_t)b * L_ + loc) * D_ + tid];
}

// ---------------------------------------------------------------------------
extern "C" void kernel_launch(void* const* d_in, const int* in_sizes, int n_in,
                              void* d_out, int out_size, void* d_ws, size_t ws_size,
                              hipStream_t stream) {
    (void)in_sizes; (void)n_in; (void)out_size; (void)ws_size;
    const float* features = (const float*)d_in[0];
    const float* hidden   = (const float*)d_in[1];
    const float* Wf       = (const float*)d_in[2];
    const float* bf       = (const float*)d_in[3];
    const float* Wh       = (const float*)d_in[4];
    const float* bh       = (const float*)d_in[5];
    const float* Ws       = (const float*)d_in[6];
    const float* bs       = (const float*)d_in[7];
    const float* noise    = (const float*)d_in[8];
    const int*   greedy   = (const int*)d_in[9];

    float* ctx   = (float*)d_out;             // [B, D] context, first output
    float* alpha = (float*)d_out + B_ * D_;   // [B, L, 1] alpha, second output
    float* cvec  = (float*)d_ws;              // [B, U] = bf + bh + hidden@Wh

    hidproj_kernel<<<dim3(B_), dim3(U_), 0, stream>>>(hidden, Wh, bf, bh, cvec);
    fused_logits_kernel<<<dim3(L_ / 128, B_), dim3(256), 0, stream>>>(
        features, Wf, cvec, Ws, bs, alpha);
    softmax_select_kernel<<<dim3(B_), dim3(256), 0, stream>>>(
        alpha, noise, greedy, features, ctx);
}
